// GATLayer_3530463117871
// MI455X (gfx1250) — compile-verified, matrix operand order checked by reading, not run on hardware
//
#include <hip/hip_runtime.h>
#include <math.h>

// ---------------------------------------------------------------------------
// GAT layer for MI455X (gfx1250, wave32, WMMA).
//   proj/skip GEMMs: bf16 WMMA 16x16x32, f32 accumulate.
//     - W pre-converted to bf16 and transposed [N][K] so B fragments are two
//       global_load_b128 per matrix per K-step.
//     - A tile staged in LDS as bf16; fragments are two ds_load_b128.
//   edge softmax + scatter-add: f32, native f32 atomics, L2-resident (<192MB).
// ---------------------------------------------------------------------------

typedef __attribute__((ext_vector_type(16))) __bf16 v16bf;
typedef __attribute__((ext_vector_type(8)))  __bf16 v8bf;
typedef __attribute__((ext_vector_type(8)))  float  v8f;

#define FIN 128
#define HEADS 8
#define FEAT 16
#define HF 128   // HEADS*FEAT

// ---------------------------------------------------------------------------
// K0: initialize accumulator (d_out), denom, and global max.
// ---------------------------------------------------------------------------
__global__ void gat_init(float* __restrict__ denom, float* __restrict__ gmax,
                         float* __restrict__ out, int n) {
  int idx = blockIdx.x * blockDim.x + threadIdx.x;
  if (idx < n * HF) out[idx] = 0.0f;
  if (idx < n * HEADS) denom[idx] = 0.0f;
  if (idx == 0) gmax[0] = -__builtin_inff();
}

// ---------------------------------------------------------------------------
// K0b: convert both weight matrices f32[K][N] -> bf16 transposed [N][K].
// Only 2 x 16K elements; runs once per launch.
// ---------------------------------------------------------------------------
__global__ void gat_wcvt(const float* __restrict__ Wp, const float* __restrict__ Ws,
                         __bf16* __restrict__ WpT, __bf16* __restrict__ WsT) {
  int idx = blockIdx.x * blockDim.x + threadIdx.x;
  if (idx >= FIN * HF) return;
  int k = idx >> 7;       // K index (row of W)
  int ncol = idx & 127;   // N index (col of W)
  WpT[ncol * FIN + k] = (__bf16)Wp[k * HF + ncol];
  WsT[ncol * FIN + k] = (__bf16)Ws[k * HF + ncol];
}

// ---------------------------------------------------------------------------
// K1: proj = x @ W_proj, skip = x @ W_skip  (16-row tile per block, 8 waves).
// Fully unrolled K loop: 8 static v_wmma_f32_16x16x32_bf16 per wave.
// ---------------------------------------------------------------------------
__global__ __launch_bounds__(256)
void gat_proj_wmma(const float* __restrict__ x,
                   const __bf16* __restrict__ WpT,   // [N][K] bf16
                   const __bf16* __restrict__ WsT,   // [N][K] bf16
                   float* __restrict__ proj,
                   float* __restrict__ skip,
                   int n) {
  // Row stride 136 bf16 = 272B: 16B-aligned rows, bank-conflict friendly.
  __shared__ __bf16 As[16][FIN + 8];

  const int t    = threadIdx.x;
  const int row0 = blockIdx.x * 16;

  // Cooperative load of the 16x128 x-tile into LDS, f32 -> bf16.
  for (int i = t; i < 16 * FIN; i += 256) {
    int r = i >> 7, c = i & 127;
    int gr = row0 + r;
    float v = (gr < n) ? x[(size_t)gr * FIN + c] : 0.0f;
    As[r][c] = (__bf16)v;
  }
  __syncthreads();

  const int wave = t >> 5;
  const int lane = t & 31;
  const int m    = lane & 15;
  const int kb   = (lane < 16) ? 0 : 8;    // 16-bit A/B lane layout (ISA 7.12.2)
  const int ncol = wave * 16 + m;

  // Fragment sources as 8xbf16 (16B) chunks.
  const v8bf* aRow  = (const v8bf*)(&As[m][0]);          // 17 chunks/row
  const v8bf* bpRow = (const v8bf*)(WpT + (size_t)ncol * FIN); // 16 chunks/row
  const v8bf* bsRow = (const v8bf*)(WsT + (size_t)ncol * FIN);

  v8f accP = {};
  v8f accS = {};

#pragma unroll
  for (int kk = 0; kk < FIN; kk += 32) {
    const int c0 = (kk + kb) >> 3;     // low chunk: K = kk+kb .. +8
    const int c1 = c0 + 2;             // high chunk: K = kk+kb+16 .. +8
    v8bf alo = aRow[c0],  ahi = aRow[c1];    // 2x ds_load_b128
    v8bf plo = bpRow[c0], phi = bpRow[c1];   // 2x global_load_b128
    v8bf slo = bsRow[c0], shi = bsRow[c1];   // 2x global_load_b128

    v16bf a  = __builtin_shufflevector(alo, ahi,
        0,1,2,3,4,5,6,7,8,9,10,11,12,13,14,15);
    v16bf bP = __builtin_shufflevector(plo, phi,
        0,1,2,3,4,5,6,7,8,9,10,11,12,13,14,15);
    v16bf bS = __builtin_shufflevector(slo, shi,
        0,1,2,3,4,5,6,7,8,9,10,11,12,13,14,15);

    accP = __builtin_amdgcn_wmma_f32_16x16x32_bf16(
        false, a, false, bP, (short)0, accP, false, false);
    accS = __builtin_amdgcn_wmma_f32_16x16x32_bf16(
        false, a, false, bS, (short)0, accS, false, false);
  }

  // C/D layout: VGPR r -> M=r (lanes 0-15), M=r+8 (lanes 16-31); N = lane&15.
  const int rbase = row0 + ((lane < 16) ? 0 : 8);
  if (row0 + 16 <= n) {
    // Full tile: unconditional stores (N % 16 == 0 path).
#pragma unroll
    for (int r = 0; r < 8; ++r) {
      proj[(size_t)(rbase + r) * HF + ncol] = accP[r];
      skip[(size_t)(rbase + r) * HF + ncol] = accS[r];
    }
  } else {
#pragma unroll
    for (int r = 0; r < 8; ++r) {
      if (rbase + r < n) {
        proj[(size_t)(rbase + r) * HF + ncol] = accP[r];
        skip[(size_t)(rbase + r) * HF + ncol] = accS[r];
      }
    }
  }
}

// ---------------------------------------------------------------------------
// K2: per-node attention halves  s_src/s_trg [N,H]
// ---------------------------------------------------------------------------
__global__ void gat_node_scores(const float* __restrict__ proj,
                                const float* __restrict__ a_src,
                                const float* __restrict__ a_trg,
                                float* __restrict__ s_src,
                                float* __restrict__ s_trg, int n) {
  int idx = blockIdx.x * blockDim.x + threadIdx.x;
  if (idx >= n * HEADS) return;
  int node = idx >> 3, h = idx & 7;
  const float4* p  = (const float4*)(proj  + (size_t)node * HF + h * FEAT);
  const float4* as = (const float4*)(a_src + h * FEAT);
  const float4* at = (const float4*)(a_trg + h * FEAT);
  float ss = 0.0f, st = 0.0f;
#pragma unroll
  for (int q = 0; q < 4; ++q) {
    float4 v = p[q], sa = as[q], ta = at[q];
    ss += v.x * sa.x + v.y * sa.y + v.z * sa.z + v.w * sa.w;
    st += v.x * ta.x + v.y * ta.y + v.z * ta.z + v.w * ta.w;
  }
  s_src[idx] = ss;
  s_trg[idx] = st;
}

__device__ __forceinline__ float leaky02(float s) {
  return (s >= 0.0f) ? s : 0.2f * s;
}

// ---------------------------------------------------------------------------
// K3: global max of leaky-relu scores (wave32 shuffle + block reduce + atomic)
// ---------------------------------------------------------------------------
__global__ __launch_bounds__(256)
void gat_edge_max(const int* __restrict__ src, const int* __restrict__ trg,
                  const float* __restrict__ s_src, const float* __restrict__ s_trg,
                  float* __restrict__ gmax, int e) {
  int idx = blockIdx.x * 256 + threadIdx.x;
  float sc = -__builtin_inff();
  if (idx < e * HEADS) {
    int edge = idx >> 3, h = idx & 7;
    sc = leaky02(s_src[src[edge] * HEADS + h] + s_trg[trg[edge] * HEADS + h]);
  }
#pragma unroll
  for (int off = 16; off > 0; off >>= 1)
    sc = fmaxf(sc, __shfl_xor(sc, off, 32));     // wave32 reduction

  __shared__ float wmaxs[8];
  if ((threadIdx.x & 31) == 0) wmaxs[threadIdx.x >> 5] = sc;
  __syncthreads();
  if (threadIdx.x < 8) {
    float v = wmaxs[threadIdx.x];
#pragma unroll
    for (int off = 4; off > 0; off >>= 1)
      v = fmaxf(v, __shfl_xor(v, off, 32));
    if (threadIdx.x == 0) {
      // float atomic max via signed-max / unsigned-min punning (init = -inf)
      if (v >= 0.0f) atomicMax((int*)gmax, __float_as_int(v));
      else           atomicMin((unsigned int*)gmax, __float_as_uint(v));
    }
  }
}

// ---------------------------------------------------------------------------
// K4: softmax denominators per (trg node, head)
// ---------------------------------------------------------------------------
__global__ void gat_edge_denom(const int* __restrict__ src, const int* __restrict__ trg,
                               const float* __restrict__ s_src, const float* __restrict__ s_trg,
                               const float* __restrict__ gmax,
                               float* __restrict__ denom, int e) {
  int idx = blockIdx.x * blockDim.x + threadIdx.x;
  if (idx >= e * HEADS) return;
  int edge = idx >> 3, h = idx & 7;
  int tn = trg[edge];
  float s  = leaky02(s_src[src[edge] * HEADS + h] + s_trg[tn * HEADS + h]);
  float ex = expf(s - gmax[0]);
  atomicAdd(&denom[tn * HEADS + h], ex);
}

// ---------------------------------------------------------------------------
// K5: weighted lift + scatter-add aggregate into accumulator (d_out)
// ---------------------------------------------------------------------------
__global__ void gat_edge_aggregate(const int* __restrict__ src, const int* __restrict__ trg,
                                   const float* __restrict__ s_src,
                                   const float* __restrict__ s_trg,
                                   const float* __restrict__ gmax,
                                   const float* __restrict__ denom,
                                   const float* __restrict__ proj,
                                   float* __restrict__ accum, int e) {
  int idx = blockIdx.x * blockDim.x + threadIdx.x;
  if (idx >= e * HEADS) return;
  int edge = idx >> 3, h = idx & 7;
  int sn = src[edge], tn = trg[edge];

  const float* p = proj + (size_t)sn * HF + h * FEAT;
  __builtin_prefetch(p, 0, 1);                 // global_prefetch_b8

  float s   = leaky02(s_src[sn * HEADS + h] + s_trg[tn * HEADS + h]);
  float att = expf(s - gmax[0]) / (denom[tn * HEADS + h] + 1e-16f);

  float* o = accum + (size_t)tn * HF + h * FEAT;
#pragma unroll
  for (int q = 0; q < 4; ++q) {
    float4 v = ((const float4*)p)[q];
    atomicAdd(o + q * 4 + 0, v.x * att);
    atomicAdd(o + q * 4 + 1, v.y * att);
    atomicAdd(o + q * 4 + 2, v.z * att);
    atomicAdd(o + q * 4 + 3, v.w * att);
  }
}

// ---------------------------------------------------------------------------
// K6: + skip + bias, ELU (in place on d_out)
// ---------------------------------------------------------------------------
__global__ void gat_epilogue(const float* __restrict__ skip,
                             const float* __restrict__ bias,
                             float* __restrict__ out, int n) {
  int idx = blockIdx.x * blockDim.x + threadIdx.x;
  if (idx >= n * HF) return;
  float v = out[idx] + skip[idx] + bias[idx & 127];
  out[idx] = (v > 0.0f) ? v : expm1f(v);
}

// ---------------------------------------------------------------------------
extern "C" void kernel_launch(void* const* d_in, const int* in_sizes, int n_in,
                              void* d_out, int out_size, void* d_ws, size_t ws_size,
                              hipStream_t stream) {
  const float* x     = (const float*)d_in[0];
  const int*   edge  = (const int*)  d_in[1];   // [2,E] int32
  const float* Wp    = (const float*)d_in[2];
  const float* a_src = (const float*)d_in[3];
  const float* a_trg = (const float*)d_in[4];
  const float* Ws    = (const float*)d_in[5];
  const float* bias  = (const float*)d_in[6];
  float*       out   = (float*)d_out;

  const int n = in_sizes[0] / FIN;
  const int e = in_sizes[1] / 2;
  const int* src = edge;
  const int* trg = edge + e;

  // workspace layout: [WpT bf16 | WsT bf16] (64KB, 16B aligned) then f32 arrays
  char* base = (char*)d_ws;
  __bf16* WpT = (__bf16*)base;                       // FIN*HF
  __bf16* WsT = WpT + (size_t)FIN * HF;              // FIN*HF
  float* proj  = (float*)(base + 2u * FIN * HF * sizeof(__bf16));
  float* skip  = proj  + (size_t)n * HF;
  float* ssrc  = skip  + (size_t)n * HF;
  float* strg  = ssrc  + (size_t)n * HEADS;
  float* denom = strg  + (size_t)n * HEADS;
  float* gmax  = denom + (size_t)n * HEADS;

  const int nhf = n * HF;       // 12.8M
  const int eh  = e * HEADS;    // 12.8M

  gat_init<<<(nhf + 255) / 256, 256, 0, stream>>>(denom, gmax, out, n);
  gat_wcvt<<<(FIN * HF + 255) / 256, 256, 0, stream>>>(Wp, Ws, WpT, WsT);
  gat_proj_wmma<<<(n + 15) / 16, 256, 0, stream>>>(x, WpT, WsT, proj, skip, n);
  gat_node_scores<<<(n * HEADS + 255) / 256, 256, 0, stream>>>(proj, a_src, a_trg,
                                                               ssrc, strg, n);
  gat_edge_max<<<(eh + 255) / 256, 256, 0, stream>>>(src, trg, ssrc, strg, gmax, e);
  gat_edge_denom<<<(eh + 255) / 256, 256, 0, stream>>>(src, trg, ssrc, strg, gmax,
                                                       denom, e);
  gat_edge_aggregate<<<(eh + 255) / 256, 256, 0, stream>>>(src, trg, ssrc, strg, gmax,
                                                           denom, proj, out, e);
  gat_epilogue<<<(nhf + 255) / 256, 256, 0, stream>>>(skip, bias, out, n);
}
